// SPF_64372969832847
// MI455X (gfx1250) — matrix operand order repaired and checked by reading, not compiled, not run on hardware
//
#include <hip/hip_runtime.h>
#include <hip/hip_bf16.h>

typedef __attribute__((ext_vector_type(16))) _Float16 v16h;
typedef __attribute__((ext_vector_type(8)))  float    v8f;

#define BATCH  4
#define NPTS   262144
#define CIN    64
#define NCH    38            // 3 + 3 + 16 + 16
#define IMG_H  544
#define IMG_W  960
#define HW     (IMG_H * IMG_W)
#define BHW    (BATCH * HW)
#define TILES32 (NPTS / 32)  // 32 points per wave iteration

// ---------------------------------------------------------------- zero
__global__ void __launch_bounds__(256)
spf_zero(float* __restrict__ out, float* __restrict__ cnt)
{
    const size_t nout = (size_t)BATCH * NCH * HW;
    const size_t ncnt = (size_t)2 * BHW;
    const size_t tid    = (size_t)blockIdx.x * blockDim.x + threadIdx.x;
    const size_t stride = (size_t)gridDim.x * blockDim.x;
    for (size_t e = tid; e < nout; e += stride) out[e] = 0.0f;
    for (size_t e = tid; e < ncnt; e += stride) cnt[e] = 0.0f;
}

// ---------------------------------------------------------------- splat + fused 1x1 conv (WMMA)
// grid = (blocksX, BATCH, 2 point sets); block = 256 (8 wave32).
// Each wave iteration handles 32 points: every lane owns one point for the
// geometry/flow scatter, then two 16-row WMMA A-tiles share the resident
// B (weight) operands: 4x v_wmma_f32_16x16x32_f16. Row->pixel broadcast for
// the feature scatter uses v_readlane (constant lane) + cndmask instead of
// ds_bpermute, so no LDS round-trips serialize the atomic stream.
__global__ void __launch_bounds__(256)
spf_splat_wmma(const float* __restrict__ pos0,  const float* __restrict__ pos1,
               const int*   __restrict__ n0,    const int*   __restrict__ n1,
               const float* __restrict__ fp0,   const float* __restrict__ fp1,
               const float* __restrict__ feat0, const float* __restrict__ feat1,
               const float* __restrict__ convw, const float* __restrict__ convb,
               float* __restrict__ out, float* __restrict__ cnt)
{
    const int s    = blockIdx.z;           // point set 0/1
    const int b    = blockIdx.y;           // batch
    const int lane = threadIdx.x & 31;     // wave32
    const int n16  = lane & 15;
    const int half = lane >> 4;
    const int wid  = blockIdx.x * (blockDim.x >> 5) + (threadIdx.x >> 5);
    const int nw   = gridDim.x * (blockDim.x >> 5);

    const float* pos  = (s == 0) ? pos0  : pos1;
    const float* fp   = (s == 0) ? fp0   : fp1;
    const float* feat = (s == 0) ? feat0 : feat1;
    const int    nv   = (s == 0) ? n0[b] : n1[b];

    const float* posb  = pos  + (size_t)b * NPTS * 3;   // [N,3]
    const float* fpb   = fp   + (size_t)b * 3 * NPTS;   // [3,N]
    const float* featb = feat + (size_t)b * CIN * NPTS; // [64,N]
    float* outb = out + (size_t)b * NCH * HW;
    float* cntb = cnt + (size_t)s * BHW + (size_t)b * HW;

    // B matrix (weights) in WMMA 16-bit B layout:
    // lane holds column n = lane%16; lanes 0-15 K=0..7,16..23, lanes 16-31 K=8..15,24..31.
    v16h bw0, bw1;
#pragma unroll
    for (int i = 0; i < 16; ++i) {
        const int K = (i < 8) ? (8 * half + i) : (16 + 8 * half + (i - 8));
        bw0[i] = (_Float16)convw[n16 * CIN + K];        // K-chunk 0..31
        bw1[i] = (_Float16)convw[n16 * CIN + 32 + K];   // K-chunk 32..63
    }
    const float bias = convb[n16];

    for (int tile = wid; tile < TILES32; tile += nw) {
        const int i0 = tile << 5;

        // ---- per-point geometry + flow-map scatter: every lane owns a point
        const int   i  = i0 + lane;
        const float fx = fpb[i]            * 4.0f;   // flow * POS_SCALE * 0.5
        const float fy = fpb[NPTS + i]     * 4.0f;
        const float fz = fpb[2 * NPTS + i] * 4.0f;
        const float x  = posb[i * 3 + 0] + fx;
        const float y  = posb[i * 3 + 1] + fy;
        const int px = (int)rintf(x);
        const int py = (int)rintf(y);
        const bool vld = (px >= 0) && (px < IMG_W) && (py >= 0) && (py < IMG_H)
                         && (i < nv);
        const int pixv = vld ? (py * IMG_W + px) : -1;   // packed pixel+valid
        if (vld) {
            unsafeAtomicAdd(&outb[(3 * s + 0) * HW + pixv], -fx);
            unsafeAtomicAdd(&outb[(3 * s + 1) * HW + pixv], -fy);
            unsafeAtomicAdd(&outb[(3 * s + 2) * HW + pixv], -fz);
            unsafeAtomicAdd(&cntb[pixv], 1.0f);
        }

        // ---- A matrices: two 16-point tiles x 64 channels, WMMA 16-bit A
        // layout. Lane holds row m = lane%16; element e -> K as below.
        const float* fcolA = featb + i0 + n16;        // points i0 .. i0+15
        const float* fcolB = fcolA + 16;              // points i0+16 .. i0+31
        v16h a0, a1, a2, a3;
#pragma unroll
        for (int e = 0; e < 16; ++e) {
            const int j = e >> 1, r = e & 1;
            const int K = ((j < 4) ? 0 : 16) + 8 * half + 2 * (j & 3) + r;
            a0[e] = (_Float16)fcolA[(size_t)K * NPTS];
            a1[e] = (_Float16)fcolA[(size_t)(K + 32) * NPTS];
            a2[e] = (_Float16)fcolB[(size_t)K * NPTS];
            a3[e] = (_Float16)fcolB[(size_t)(K + 32) * NPTS];
        }

        if (tile + nw < TILES32)                      // global_prefetch_b8
            __builtin_prefetch(fcolA + (size_t)nw * 32, 0, 1);

        // ---- pf = A x Wt + bias  (fp32 accum, K = 64 via 2 chained WMMAs each)
        v8f cA, cB;
#pragma unroll
        for (int r = 0; r < 8; ++r) { cA[r] = bias; cB[r] = bias; }
        cA = __builtin_amdgcn_wmma_f32_16x16x32_f16(false, a0, false, bw0,
                                                    (short)0, cA, false, false);
        cB = __builtin_amdgcn_wmma_f32_16x16x32_f16(false, a2, false, bw0,
                                                    (short)0, cB, false, false);
        cA = __builtin_amdgcn_wmma_f32_16x16x32_f16(false, a1, false, bw1,
                                                    (short)0, cA, false, false);
        cB = __builtin_amdgcn_wmma_f32_16x16x32_f16(false, a3, false, bw1,
                                                    (short)0, cB, false, false);

        // ---- row -> pixel broadcast without LDS: source lane is a constant
        // per half (r + 8*half, +16 for tile B), so two v_readlane + one
        // cndmask per row replace ds_bpermute + dscnt waits.
        int pmA[8], pmB[8];
#pragma unroll
        for (int r = 0; r < 8; ++r) {
            const int aLo = __builtin_amdgcn_readlane(pixv, r);       // row r
            const int aHi = __builtin_amdgcn_readlane(pixv, r + 8);   // row r+8
            const int bLo = __builtin_amdgcn_readlane(pixv, r + 16);
            const int bHi = __builtin_amdgcn_readlane(pixv, r + 24);
            pmA[r] = half ? aHi : aLo;
            pmB[r] = half ? bHi : bLo;
        }

        // ---- feature-map scatter: lane owns out-channel n16; accumulator
        // VGPR r holds row M = r + 8*half (C/D layout). pm < 0 == invalid.
        const int fchan = (6 + 16 * s + n16) * HW;
#pragma unroll
        for (int r = 0; r < 8; ++r) {
            if (pmA[r] >= 0) unsafeAtomicAdd(&outb[fchan + pmA[r]], cA[r]);
            if (pmB[r] >= 0) unsafeAtomicAdd(&outb[fchan + pmB[r]], cB[r]);
        }
    }
}

// ---------------------------------------------------------------- divide by counts
__global__ void __launch_bounds__(256)
spf_finalize(float* __restrict__ out, const float* __restrict__ cnt)
{
    const size_t total  = (size_t)BATCH * NCH * HW;
    const size_t tid    = (size_t)blockIdx.x * blockDim.x + threadIdx.x;
    const size_t stride = (size_t)gridDim.x * blockDim.x;
    for (size_t e = tid; e < total; e += stride) {
        const size_t pix = e % HW;
        const size_t bc  = e / HW;
        const int c = (int)(bc % NCH);
        const int b = (int)(bc / NCH);
        const int s = (c < 3) ? 0 : (c < 6) ? 1 : (c < 22) ? 0 : 1;
        const float cv = cnt[(size_t)s * BHW + (size_t)b * HW + pix];
        out[e] = out[e] / fmaxf(cv, 1.0f);
    }
}

extern "C" void kernel_launch(void* const* d_in, const int* in_sizes, int n_in,
                              void* d_out, int out_size, void* d_ws, size_t ws_size,
                              hipStream_t stream)
{
    const float* pos0  = (const float*)d_in[0];
    const float* pos1  = (const float*)d_in[1];
    const int*   n0    = (const int*)  d_in[2];
    const int*   n1    = (const int*)  d_in[3];
    // d_in[4..6] = img0, img1, embt: unused by the rasterization math
    const float* fp0   = (const float*)d_in[7];
    const float* fp1   = (const float*)d_in[8];
    const float* feat0 = (const float*)d_in[9];
    const float* feat1 = (const float*)d_in[10];
    const float* convw = (const float*)d_in[11];
    const float* convb = (const float*)d_in[12];

    float* out = (float*)d_out;
    float* cnt = (float*)d_ws;   // 2 * BATCH * HW floats

    spf_zero<<<2048, 256, 0, stream>>>(out, cnt);

    dim3 grid(256, BATCH, 2);    // 2048 waves per (b,s); TILES32=8192 -> 4 iters/wave
    spf_splat_wmma<<<grid, 256, 0, stream>>>(pos0, pos1, n0, n1, fp0, fp1,
                                             feat0, feat1, convw, convb, out, cnt);

    spf_finalize<<<4096, 256, 0, stream>>>(out, cnt);
}
